// STM_GrootV3D_V2_46978352283922
// MI455X (gfx1250) — compile-verified, matrix operand order checked by reading, not run on hardware
//
#include <hip/hip_runtime.h>

// ---------------- shapes ----------------
#define L_SEQ   4608          // 48 * 96
#define C_DIM   640
#define DI      1280
#define DSTATE  8
#define NXZ     2560          // 2*DI
#define NPROJ   64            // 56 padded to 64
#define KDT     64            // DT_RANK=40 padded to 64
#define NC      64            // scan chunks
#define LC      72            // steps per chunk (64*72 = 4608)
#define HW      2304          // 48*48
#define XYOFF   1474560       // 640*2304

// ---------------- types ----------------
typedef __bf16  bf16_t;
typedef bf16_t  v16bf __attribute__((ext_vector_type(16)));
typedef bf16_t  v8bf  __attribute__((ext_vector_type(8)));
typedef float   v8f   __attribute__((ext_vector_type(8)));

union fvec { v8f v; float f[8]; };

static __device__ inline unsigned short f2bf(float f) {
    union { float f; unsigned int u; } v; v.f = f;
    unsigned int u = v.u;
    unsigned int r = u + 0x7FFFu + ((u >> 16) & 1u);   // round-to-nearest-even
    return (unsigned short)(r >> 16);
}

// Load one 16x(K=32) bf16 WMMA fragment row (A: row=M, B: row=N of B^T).
// Wave32 layout: lanes 0-15 hold K = {0..7, 16..23}, lanes 16-31 hold K = {8..15, 24..31}.
static __device__ inline v16bf load_frag(const unsigned short* base, int row, int ld,
                                         int kk, int half) {
    const unsigned short* p = base + (size_t)row * ld + kk + 8 * half;
    v8bf lo = *(const v8bf*)(p);
    v8bf hi = *(const v8bf*)(p + 16);
    return __builtin_shufflevector(lo, hi, 0,1,2,3,4,5,6,7,8,9,10,11,12,13,14,15);
}

static __device__ inline void store_tile(float* C, int ldc, const fvec& acc,
                                         int mb, int nb, int r, int half) {
    int n = nb + r;
    #pragma unroll
    for (int i = 0; i < 8; ++i) {
        int m = mb + i + 8 * half;
        C[(size_t)m * ldc + n] = acc.f[i];
    }
}

// ---------------- weight prep: f32 (R x Cc) -> bf16 transposed+padded (Nd x Kd) ----------------
__global__ void transpose_pad_bf_kernel(const float* __restrict__ src, unsigned short* __restrict__ dst,
                                        int R, int Cc, int Nd, int Kd) {
    int i = blockIdx.x * blockDim.x + threadIdx.x;
    if (i >= Nd * Kd) return;
    int n = i / Kd, k = i - n * Kd;
    dst[i] = (n < Cc && k < R) ? f2bf(src[(size_t)k * Cc + n]) : (unsigned short)0;
}

__global__ void aneg_kernel(const float* __restrict__ A_log, float* __restrict__ Aneg) {
    int i = blockIdx.x * blockDim.x + threadIdx.x;
    if (i < DI * DSTATE) Aneg[i] = -__expf(A_log[i]);
}

// ---------------- gather (x|y interleave) + LayerNorm ----------------
__global__ void gather_ln_kernel(const float* __restrict__ x, const float* __restrict__ y,
                                 const float* __restrict__ g, const float* __restrict__ b,
                                 float* __restrict__ seq_raw, unsigned short* __restrict__ xn_bf) {
    int l = blockIdx.x;                       // 0..4607
    int h = l / 96, w2 = l - h * 96;
    const float* src = (w2 < 48) ? (x + h * 48 + w2) : (y + h * 48 + (w2 - 48));
    int tid = threadIdx.x;

    __shared__ float rs[256], rq[256];
    float vals[3];
    float s = 0.f, s2 = 0.f;
    int cnt = 0;
    for (int c = tid; c < C_DIM; c += 256) {
        float v = src[(size_t)c * HW];
        vals[cnt++] = v; s += v; s2 += v * v;
    }
    rs[tid] = s; rq[tid] = s2;
    __syncthreads();
    for (int st = 128; st > 0; st >>= 1) {
        if (tid < st) { rs[tid] += rs[tid + st]; rq[tid] += rq[tid + st]; }
        __syncthreads();
    }
    float mean = rs[0] * (1.f / C_DIM);
    float var  = rq[0] * (1.f / C_DIM) - mean * mean;
    float inv  = rsqrtf(var + 1e-5f);

    cnt = 0;
    for (int c = tid; c < C_DIM; c += 256) {
        float v = vals[cnt++];
        seq_raw[(size_t)l * C_DIM + c] = v;
        float nv = (v - mean) * inv * g[c] + b[c];
        xn_bf[(size_t)l * C_DIM + c] = f2bf(nv);
    }
}

// ---------------- generic bf16 WMMA GEMM: C(MxN) = A(MxK) * Bt(NxK)^T ----------------
__global__ void gemm_bf16_kernel(const unsigned short* __restrict__ A,
                                 const unsigned short* __restrict__ Bt,
                                 float* __restrict__ C, int M, int N, int K) {
    int wid  = (blockIdx.x * blockDim.x + threadIdx.x) >> 5;
    int lane = threadIdx.x & 31;
    int r = lane & 15, half = lane >> 4;
    int ntiles = N >> 5;
    int total  = (M >> 5) * ntiles;
    if (wid >= total) return;                 // wave-uniform: EXEC stays all-ones for WMMA
    int tm = wid / ntiles, tn = wid - tm * ntiles;
    int m0 = tm * 32, n0 = tn * 32;

    fvec acc00 = {}, acc01 = {}, acc10 = {}, acc11 = {};
    for (int kk = 0; kk < K; kk += 32) {
        v16bf a0 = load_frag(A,  m0 + r,      K, kk, half);
        v16bf a1 = load_frag(A,  m0 + 16 + r, K, kk, half);
        v16bf b0 = load_frag(Bt, n0 + r,      K, kk, half);
        v16bf b1 = load_frag(Bt, n0 + 16 + r, K, kk, half);
        acc00.v = __builtin_amdgcn_wmma_f32_16x16x32_bf16(false, a0, false, b0, (short)0, acc00.v, false, false);
        acc01.v = __builtin_amdgcn_wmma_f32_16x16x32_bf16(false, a0, false, b1, (short)0, acc01.v, false, false);
        acc10.v = __builtin_amdgcn_wmma_f32_16x16x32_bf16(false, a1, false, b0, (short)0, acc10.v, false, false);
        acc11.v = __builtin_amdgcn_wmma_f32_16x16x32_bf16(false, a1, false, b1, (short)0, acc11.v, false, false);
    }
    store_tile(C, N, acc00, m0,      n0,      r, half);
    store_tile(C, N, acc01, m0,      n0 + 16, r, half);
    store_tile(C, N, acc10, m0 + 16, n0,      r, half);
    store_tile(C, N, acc11, m0 + 16, n0 + 16, r, half);
}

// ---------------- depthwise causal conv (K=4) + SiLU ----------------
__global__ void conv_silu_kernel(const float* __restrict__ xz, const float* __restrict__ cw,
                                 float* __restrict__ xa, unsigned short* __restrict__ xa_bf) {
    int idx = blockIdx.x * blockDim.x + threadIdx.x;
    if (idx >= L_SEQ * DI) return;
    int l = idx / DI, d = idx - l * DI;
    float acc = 0.f;
    #pragma unroll
    for (int k = 0; k < 4; ++k) {
        int ls = l - 3 + k;
        if (ls >= 0) acc += cw[d * 4 + k] * xz[(size_t)ls * NXZ + d];   // xp = xz[:, :DI]
    }
    float sv = acc / (1.f + __expf(-acc));
    xa[idx]    = sv;
    xa_bf[idx] = f2bf(sv);
}

// ---------------- split proj -> (dt input padded bf16, B, C) ----------------
__global__ void split_proj_kernel(const float* __restrict__ proj, unsigned short* __restrict__ dtin_bf,
                                  float* __restrict__ Bm, float* __restrict__ Cm) {
    int idx = blockIdx.x * blockDim.x + threadIdx.x;
    if (idx >= L_SEQ * NPROJ) return;
    int l = idx / NPROJ, c = idx - l * NPROJ;
    float v = proj[idx];
    dtin_bf[idx] = (c < 40) ? f2bf(v) : (unsigned short)0;
    if (c >= 40 && c < 48)      Bm[l * DSTATE + (c - 40)] = v;
    else if (c >= 48 && c < 56) Cm[l * DSTATE + (c - 48)] = v;
}

// ---------------- dt = softplus(dtraw + bias) ----------------
__global__ void dt_kernel(const float* __restrict__ dtraw, const float* __restrict__ dt_bias,
                          float* __restrict__ dt) {
    int idx = blockIdx.x * blockDim.x + threadIdx.x;
    if (idx >= L_SEQ * DI) return;
    int d = idx % DI;
    float v = dtraw[idx] + dt_bias[d];
    dt[idx] = (v > 20.f) ? v : log1pf(__expf(v));
}

// ---------------- scan phase 1: per-chunk local scan ----------------
__global__ void scan1_kernel(const float* __restrict__ dt, const float* __restrict__ xa,
                             const float* __restrict__ Bm, const float* __restrict__ Aneg,
                             float* __restrict__ Aprod, float* __restrict__ Hloc) {
    int idx = blockIdx.x * blockDim.x + threadIdx.x;   // NC*DI
    if (idx >= NC * DI) return;
    int chunk = idx / DI, d = idx - chunk * DI;
    float Ad[DSTATE], h[DSTATE], ap[DSTATE];
    #pragma unroll
    for (int s = 0; s < DSTATE; ++s) { Ad[s] = Aneg[d * DSTATE + s]; h[s] = 0.f; ap[s] = 1.f; }
    int l0 = chunk * LC;
    for (int t = 0; t < LC; ++t) {
        int l = l0 + t;
        float dtv = dt[(size_t)l * DI + d];
        float xav = xa[(size_t)l * DI + d];
        float w = dtv * xav;
        #pragma unroll
        for (int s = 0; s < DSTATE; ++s) {
            float da = __expf(dtv * Ad[s]);
            h[s]  = da * h[s] + w * Bm[l * DSTATE + s];
            ap[s] = ap[s] * da;
        }
    }
    size_t o = (size_t)idx * DSTATE;
    #pragma unroll
    for (int s = 0; s < DSTATE; ++s) { Aprod[o + s] = ap[s]; Hloc[o + s] = h[s]; }
}

// ---------------- scan phase 2: cross-chunk combine ----------------
__global__ void scan2_kernel(const float* __restrict__ Aprod, const float* __restrict__ Hloc,
                             float* __restrict__ Hinit) {
    int idx = blockIdx.x * blockDim.x + threadIdx.x;   // DI*DSTATE
    if (idx >= DI * DSTATE) return;
    float h = 0.f;
    for (int c = 0; c < NC; ++c) {
        size_t o = (size_t)c * DI * DSTATE + idx;
        Hinit[o] = h;
        h = Aprod[o] * h + Hloc[o];
    }
}

// ---------------- scan phase 3: re-scan with init state, fuse C-contract + D + gate ----------------
__global__ void scan3_kernel(const float* __restrict__ dt, const float* __restrict__ xa,
                             const float* __restrict__ Bm, const float* __restrict__ Cm,
                             const float* __restrict__ Aneg, const float* __restrict__ Dp,
                             const float* __restrict__ xz, const float* __restrict__ Hinit,
                             unsigned short* __restrict__ yo_bf) {
    int idx = blockIdx.x * blockDim.x + threadIdx.x;   // NC*DI
    if (idx >= NC * DI) return;
    int chunk = idx / DI, d = idx - chunk * DI;
    float Ad[DSTATE], h[DSTATE];
    size_t o = (size_t)idx * DSTATE;
    #pragma unroll
    for (int s = 0; s < DSTATE; ++s) { Ad[s] = Aneg[d * DSTATE + s]; h[s] = Hinit[o + s]; }
    float Dv = Dp[d];
    int l0 = chunk * LC;
    for (int t = 0; t < LC; ++t) {
        int l = l0 + t;
        float dtv = dt[(size_t)l * DI + d];
        float xav = xa[(size_t)l * DI + d];
        float w = dtv * xav;
        float ys = 0.f;
        #pragma unroll
        for (int s = 0; s < DSTATE; ++s) {
            float da = __expf(dtv * Ad[s]);
            h[s] = da * h[s] + w * Bm[l * DSTATE + s];
            ys += Cm[l * DSTATE + s] * h[s];
        }
        ys += xav * Dv;
        float zv = xz[(size_t)l * NXZ + DI + d];       // z = xz[:, DI:]
        float yo = ys * (zv / (1.f + __expf(-zv)));
        yo_bf[(size_t)l * DI + d] = f2bf(yo);
    }
}

// ---------------- final GEMM + residual + scatter to (x_out | y_out) ----------------
static __device__ inline void emit_out(const fvec& acc, int mb, int nb, int r, int half,
                                       const float* __restrict__ seq_raw, float* __restrict__ out) {
    int c = nb + r;
    #pragma unroll
    for (int i = 0; i < 8; ++i) {
        int l = mb + i + 8 * half;
        float v = acc.f[i] + seq_raw[(size_t)l * C_DIM + c];
        int hh = l / 96, w2 = l - hh * 96;
        int dst = (w2 < 48) ? (c * HW + hh * 48 + w2)
                            : (XYOFF + c * HW + hh * 48 + (w2 - 48));
        out[dst] = v;
    }
}

__global__ void gemm_out_kernel(const unsigned short* __restrict__ A,
                                const unsigned short* __restrict__ Bt,
                                const float* __restrict__ seq_raw, float* __restrict__ out) {
    const int M = L_SEQ, N = C_DIM, K = DI;
    int wid  = (blockIdx.x * blockDim.x + threadIdx.x) >> 5;
    int lane = threadIdx.x & 31;
    int r = lane & 15, half = lane >> 4;
    int ntiles = N >> 5;
    int total  = (M >> 5) * ntiles;
    if (wid >= total) return;
    int tm = wid / ntiles, tn = wid - tm * ntiles;
    int m0 = tm * 32, n0 = tn * 32;

    fvec acc00 = {}, acc01 = {}, acc10 = {}, acc11 = {};
    for (int kk = 0; kk < K; kk += 32) {
        v16bf a0 = load_frag(A,  m0 + r,      K, kk, half);
        v16bf a1 = load_frag(A,  m0 + 16 + r, K, kk, half);
        v16bf b0 = load_frag(Bt, n0 + r,      K, kk, half);
        v16bf b1 = load_frag(Bt, n0 + 16 + r, K, kk, half);
        acc00.v = __builtin_amdgcn_wmma_f32_16x16x32_bf16(false, a0, false, b0, (short)0, acc00.v, false, false);
        acc01.v = __builtin_amdgcn_wmma_f32_16x16x32_bf16(false, a0, false, b1, (short)0, acc01.v, false, false);
        acc10.v = __builtin_amdgcn_wmma_f32_16x16x32_bf16(false, a1, false, b0, (short)0, acc10.v, false, false);
        acc11.v = __builtin_amdgcn_wmma_f32_16x16x32_bf16(false, a1, false, b1, (short)0, acc11.v, false, false);
    }
    emit_out(acc00, m0,      n0,      r, half, seq_raw, out);
    emit_out(acc01, m0,      n0 + 16, r, half, seq_raw, out);
    emit_out(acc10, m0 + 16, n0,      r, half, seq_raw, out);
    emit_out(acc11, m0 + 16, n0 + 16, r, half, seq_raw, out);
}

// ---------------- host launcher ----------------
extern "C" void kernel_launch(void* const* d_in, const int* in_sizes, int n_in,
                              void* d_out, int out_size, void* d_ws, size_t ws_size,
                              hipStream_t stream) {
    const float* x       = (const float*)d_in[0];
    const float* y       = (const float*)d_in[1];
    const float* ln_g    = (const float*)d_in[2];
    const float* ln_b    = (const float*)d_in[3];
    const float* W_in    = (const float*)d_in[4];
    const float* conv_w  = (const float*)d_in[5];
    const float* W_xproj = (const float*)d_in[6];
    const float* W_dt    = (const float*)d_in[7];
    const float* dt_bias = (const float*)d_in[8];
    const float* A_log   = (const float*)d_in[9];
    const float* Dp      = (const float*)d_in[10];
    const float* W_out   = (const float*)d_in[11];
    float* out = (float*)d_out;

    char* ws = (char*)d_ws;
    size_t off = 0;
    auto alloc = [&](size_t bytes) -> size_t {
        size_t o = off;
        off += (bytes + 255) & ~(size_t)255;
        return o;
    };

    size_t o_seq    = alloc((size_t)L_SEQ * C_DIM * 4);     // seq_raw f32
    size_t o_xn     = alloc((size_t)L_SEQ * C_DIM * 2);     // xn bf16
    size_t o_WinT   = alloc((size_t)NXZ * C_DIM * 2);       // W_in^T bf16 (2560 x 640)
    size_t o_WxpT   = alloc((size_t)NPROJ * DI * 2);        // W_xproj^T bf16 (64 x 1280)
    size_t o_WdtT   = alloc((size_t)DI * KDT * 2);          // W_dt^T bf16 (1280 x 64)
    size_t o_WoutT  = alloc((size_t)C_DIM * DI * 2);        // W_out^T bf16 (640 x 1280)
    size_t o_Aneg   = alloc((size_t)DI * DSTATE * 4);
    size_t o_xz     = alloc((size_t)L_SEQ * NXZ * 4);       // xz f32
    size_t o_xa     = alloc((size_t)L_SEQ * DI * 4);
    size_t o_xabf   = alloc((size_t)L_SEQ * DI * 2);
    size_t o_proj   = alloc((size_t)L_SEQ * NPROJ * 4);
    size_t o_dtin   = alloc((size_t)L_SEQ * NPROJ * 2);
    size_t o_Bm     = alloc((size_t)L_SEQ * DSTATE * 4);
    size_t o_Cm     = alloc((size_t)L_SEQ * DSTATE * 4);
    size_t o_dtraw  = alloc((size_t)L_SEQ * DI * 4);
    size_t o_dt     = alloc((size_t)L_SEQ * DI * 4);
    size_t o_Aprod  = alloc((size_t)NC * DI * DSTATE * 4);
    size_t o_Hloc   = alloc((size_t)NC * DI * DSTATE * 4);
    size_t o_Hinit  = alloc((size_t)NC * DI * DSTATE * 4);
    size_t o_yobf   = alloc((size_t)L_SEQ * DI * 2);
    (void)ws_size; (void)in_sizes; (void)n_in; (void)out_size;

    float*          seq_raw = (float*)(ws + o_seq);
    unsigned short* xn_bf   = (unsigned short*)(ws + o_xn);
    unsigned short* WinT    = (unsigned short*)(ws + o_WinT);
    unsigned short* WxpT    = (unsigned short*)(ws + o_WxpT);
    unsigned short* WdtT    = (unsigned short*)(ws + o_WdtT);
    unsigned short* WoutT   = (unsigned short*)(ws + o_WoutT);
    float*          Aneg    = (float*)(ws + o_Aneg);
    float*          xz      = (float*)(ws + o_xz);
    float*          xa      = (float*)(ws + o_xa);
    unsigned short* xa_bf   = (unsigned short*)(ws + o_xabf);
    float*          proj    = (float*)(ws + o_proj);
    unsigned short* dtin_bf = (unsigned short*)(ws + o_dtin);
    float*          Bm      = (float*)(ws + o_Bm);
    float*          Cm      = (float*)(ws + o_Cm);
    float*          dtraw   = (float*)(ws + o_dtraw);
    float*          dt      = (float*)(ws + o_dt);
    float*          Aprod   = (float*)(ws + o_Aprod);
    float*          Hloc    = (float*)(ws + o_Hloc);
    float*          Hinit   = (float*)(ws + o_Hinit);
    unsigned short* yo_bf   = (unsigned short*)(ws + o_yobf);

    const int T = 256;
    auto blks = [](long long n, int t) { return (int)((n + t - 1) / t); };

    // weight prep
    transpose_pad_bf_kernel<<<blks((long long)NXZ * C_DIM, T), T, 0, stream>>>(W_in,    WinT,  C_DIM, NXZ,   NXZ,   C_DIM);
    transpose_pad_bf_kernel<<<blks((long long)NPROJ * DI, T),  T, 0, stream>>>(W_xproj, WxpT,  DI,    56,    NPROJ, DI);
    transpose_pad_bf_kernel<<<blks((long long)DI * KDT, T),    T, 0, stream>>>(W_dt,    WdtT,  40,    DI,    DI,    KDT);
    transpose_pad_bf_kernel<<<blks((long long)C_DIM * DI, T),  T, 0, stream>>>(W_out,   WoutT, DI,    C_DIM, C_DIM, DI);
    aneg_kernel<<<blks(DI * DSTATE, T), T, 0, stream>>>(A_log, Aneg);

    // gather + layernorm
    gather_ln_kernel<<<L_SEQ, 256, 0, stream>>>(x, y, ln_g, ln_b, seq_raw, xn_bf);

    // GEMM1: xz = xn @ W_in   (4608 x 640 x 2560)
    {
        int waves = (L_SEQ / 32) * (NXZ / 32);
        gemm_bf16_kernel<<<blks((long long)waves * 32, T), T, 0, stream>>>(xn_bf, WinT, xz, L_SEQ, NXZ, C_DIM);
    }

    // depthwise conv + SiLU
    conv_silu_kernel<<<blks((long long)L_SEQ * DI, T), T, 0, stream>>>(xz, conv_w, xa, xa_bf);

    // GEMM2: proj = xa @ W_xproj   (4608 x 1280 x 64pad)
    {
        int waves = (L_SEQ / 32) * (NPROJ / 32);
        gemm_bf16_kernel<<<blks((long long)waves * 32, T), T, 0, stream>>>(xa_bf, WxpT, proj, L_SEQ, NPROJ, DI);
    }

    split_proj_kernel<<<blks((long long)L_SEQ * NPROJ, T), T, 0, stream>>>(proj, dtin_bf, Bm, Cm);

    // GEMM3: dtraw = dt_in @ W_dt   (4608 x 64pad x 1280)
    {
        int waves = (L_SEQ / 32) * (DI / 32);
        gemm_bf16_kernel<<<blks((long long)waves * 32, T), T, 0, stream>>>(dtin_bf, WdtT, dtraw, L_SEQ, DI, KDT);
    }

    dt_kernel<<<blks((long long)L_SEQ * DI, T), T, 0, stream>>>(dtraw, dt_bias, dt);

    // chunked selective scan
    scan1_kernel<<<blks((long long)NC * DI, T), T, 0, stream>>>(dt, xa, Bm, Aneg, Aprod, Hloc);
    scan2_kernel<<<blks((long long)DI * DSTATE, T), T, 0, stream>>>(Aprod, Hloc, Hinit);
    scan3_kernel<<<blks((long long)NC * DI, T), T, 0, stream>>>(dt, xa, Bm, Cm, Aneg, Dp, xz, Hinit, yo_bf);

    // GEMM4 + residual + scatter
    {
        int waves = (L_SEQ / 32) * (C_DIM / 32);
        gemm_out_kernel<<<blks((long long)waves * 32, T), T, 0, stream>>>(yo_bf, WoutT, seq_raw, out);
    }
}
